// LlamaDecoderLayer_27917287424723
// MI455X (gfx1250) — matrix operand order repaired and checked by reading, not compiled, not run on hardware
//
#include <hip/hip_runtime.h>

// ---------------------------------------------------------------------------
// Types
// ---------------------------------------------------------------------------
typedef __bf16 bf16_t;
typedef __bf16 v8bf  __attribute__((ext_vector_type(8)));
typedef __bf16 v16bf __attribute__((ext_vector_type(16)));
typedef float  v8f   __attribute__((ext_vector_type(8)));
typedef unsigned uint32x4 __attribute__((ext_vector_type(4)));
typedef int      int32x4  __attribute__((ext_vector_type(4)));
typedef int      int32x8  __attribute__((ext_vector_type(8)));

#define T_SEQ 2048
#define D_MODEL 2048
#define FF_DIM 8192
#define KV_DIM 512
#define N_HEAD 32
#define HD 64

// float -> bf16, round-to-nearest-even
__device__ __forceinline__ bf16_t f2bf(float f) {
  union { float f; unsigned u; } a; a.f = f;
  unsigned r = a.u + 0x7FFFu + ((a.u >> 16) & 1u);
  union { unsigned short s; bf16_t b; } o; o.s = (unsigned short)(r >> 16);
  return o.b;
}

// Assemble a 16-element bf16 WMMA fragment from two 8-element (16B) chunks.
__device__ __forceinline__ v16bf pack16(const bf16_t* lo, const bf16_t* hi) {
  v8bf a = *(const v8bf*)lo;
  v8bf b = *(const v8bf*)hi;
  v16bf r;
#pragma unroll
  for (int j = 0; j < 8; ++j) { r[j] = a[j]; r[j + 8] = b[j]; }
  return r;
}

__device__ __forceinline__ v8f wmma_bf16(v16bf a, v16bf b, v8f c) {
  return __builtin_amdgcn_wmma_f32_16x16x32_bf16(false, a, false, b,
                                                 (short)0, c, false, false);
}

// ---------------------------------------------------------------------------
// Tensor Data Mover: 2D bf16 tile (tile_d0 x tile_d1) from global -> LDS.
// D# per CDNA5 ISA ch.8: group0 = {count, lds_addr, global_addr, type=2},
// group1 = {data_size=2B, tensor dims, tile dims, dim0 stride}.
// ---------------------------------------------------------------------------
__device__ __forceinline__ void tdm_load_2d_bf16(unsigned lds_addr,
                                                 const void* gptr,
                                                 unsigned tensor_d0,
                                                 unsigned tensor_d1,
                                                 unsigned tile_d0,
                                                 unsigned tile_d1,
                                                 unsigned stride0) {
  unsigned long long ga = (unsigned long long)gptr;
  uint32x4 g0;
  g0[0] = 1u;                                            // count=1, user mode
  g0[1] = lds_addr;                                      // LDS byte address
  g0[2] = (unsigned)(ga & 0xFFFFFFFFu);                  // global_addr[31:0]
  g0[3] = (unsigned)((ga >> 32) & 0x1FFFFFFu) | (2u << 30);  // addr[56:32]|type=2
  int32x8 g1;
  g1[0] = (int)(1u << 16);                               // data_size=1 (2 bytes)
  g1[1] = (int)((tensor_d0 & 0xFFFFu) << 16);            // tensor_dim0[15:0]
  g1[2] = (int)((tensor_d0 >> 16) | ((tensor_d1 & 0xFFFFu) << 16));
  g1[3] = (int)((tensor_d1 >> 16) | (tile_d0 << 16));    // tile_dim0
  g1[4] = (int)(tile_d1 & 0xFFFFu);                      // tile_dim1 (tile_dim2=0)
  g1[5] = (int)stride0;                                  // tensor_dim0_stride[31:0]
  g1[6] = 0;
  g1[7] = 0;
  int32x4 z4 = {0, 0, 0, 0};
#if defined(__clang_major__) && (__clang_major__ >= 23)
  int32x8 z8 = {0, 0, 0, 0, 0, 0, 0, 0};
  __builtin_amdgcn_tensor_load_to_lds(g0, g1, z4, z4, z8, 0);
#else
  __builtin_amdgcn_tensor_load_to_lds(g0, g1, z4, z4, 0);
#endif
}

// Per-lane async 16-byte copy global -> LDS (ASYNCcnt tracked).
__device__ __forceinline__ void async_copy_b128(unsigned lds_addr,
                                                unsigned long long gaddr) {
  asm volatile("global_load_async_to_lds_b128 %0, %1, off"
               :: "v"(lds_addr), "v"(gaddr) : "memory");
}

// ---------------------------------------------------------------------------
// RMSNorm: one block per row; writes f32 (residual path) and bf16 (WMMA path)
// ---------------------------------------------------------------------------
__global__ __launch_bounds__(256) void rmsnorm_kernel(
    const float* __restrict__ x, const float* __restrict__ g,
    float* __restrict__ outF, bf16_t* __restrict__ outH, int D) {
  __shared__ float red[8];
  __shared__ float tot;
  const int row = blockIdx.x;
  const int tid = threadIdx.x;
  const size_t base = (size_t)row * D;

  float p = 0.f;
  for (int i = tid; i < D; i += 256) { float v = x[base + i]; p += v * v; }
#pragma unroll
  for (int m = 1; m < 32; m <<= 1) p += __shfl_xor(p, m, 32);
  if ((tid & 31) == 0) red[tid >> 5] = p;
  __syncthreads();
  if (tid == 0) {
    float s = 0.f;
#pragma unroll
    for (int i = 0; i < 8; ++i) s += red[i];
    tot = rsqrtf(s / (float)D + 1e-5f);
  }
  __syncthreads();
  const float sc = tot;
  for (int i = tid; i < D; i += 256) {
    float v = g[i] * x[base + i] * sc;
    outF[base + i] = v;
    outH[base + i] = f2bf(v);
  }
}

// ---------------------------------------------------------------------------
// Weight prep: W f32 [K][N] -> Wt bf16 [N][K] (transpose via LDS tile).
// Makes every GEMM B-fragment two contiguous 16B global loads.
// ---------------------------------------------------------------------------
__global__ __launch_bounds__(256) void convert_transpose_kernel(
    const float* __restrict__ W, bf16_t* __restrict__ Wt, int K, int N) {
  __shared__ float tile[32][33];
  const int n0 = blockIdx.x * 32, k0 = blockIdx.y * 32;
  const int lx = threadIdx.x & 31, ly = threadIdx.x >> 5;
#pragma unroll
  for (int r = 0; r < 4; ++r)
    tile[ly + 8 * r][lx] = W[(size_t)(k0 + ly + 8 * r) * N + n0 + lx];
  __syncthreads();
#pragma unroll
  for (int r = 0; r < 4; ++r)
    Wt[(size_t)(n0 + ly + 8 * r) * K + k0 + lx] = f2bf(tile[lx][ly + 8 * r]);
}

// ---------------------------------------------------------------------------
// bf16 WMMA GEMM: C[M,N] = A[M,K] @ Bt[N,K]^T
// 8 waves, block tile 128x64, wave tile 32x32, K-step 64.
// A tile DMA'd by the Tensor Data Mover (double-buffered, overlapped with
// WMMA); B fragments stream straight from L2 (transposed bf16 weights).
// EPI: 0 = f32  | 2 = f32 + res1 | 3 = dual-B silu-gate -> bf16 | 4 = f32+res1+res2
// ---------------------------------------------------------------------------
template <int EPI>
__global__ __launch_bounds__(256) void gemm_kernel(
    const bf16_t* __restrict__ A, const bf16_t* __restrict__ Bt,
    const bf16_t* __restrict__ B2t, float* __restrict__ outF,
    bf16_t* __restrict__ outH, const float* __restrict__ res1,
    const float* __restrict__ res2, int M, int N, int K) {
  __shared__ __align__(128) bf16_t As[2][128][64];

  const int tid = threadIdx.x;
  const int wave = tid >> 5, lane = tid & 31;
  const int wm = wave >> 1, wn = wave & 1;
  const int ln = lane & 15;
  const int kb8 = (lane >> 4) * 8;    // unused (A fragments also use kb16)
  const int kb16 = (lane >> 4) * 16;  // B-fragment K base
  const int bm = blockIdx.y, bn = blockIdx.x;

  v8f acc[2][2] = {};
  v8f acc2[2][2] = {};

  const bf16_t* Abase = A + (size_t)bm * 128 * K;

  if (wave == 0)
    tdm_load_2d_bf16((unsigned)(size_t)&As[0][0][0], Abase,
                     (unsigned)K, (unsigned)M, 64u, 128u, (unsigned)K);

  for (int k0 = 0; k0 < K; k0 += 64) {
    const int buf = (k0 >> 6) & 1;
    if (wave == 0) __builtin_amdgcn_s_wait_tensorcnt(0);
    __syncthreads();  // TDM tile visible to all; prev buffer fully consumed
    if (wave == 0 && (k0 + 64) < K)
      tdm_load_2d_bf16((unsigned)(size_t)&As[buf ^ 1][0][0], Abase + k0 + 64,
                       (unsigned)K, (unsigned)M, 64u, 128u, (unsigned)K);
    if ((k0 + 64) < K)  // warm L2 for next B stripe
      __builtin_prefetch(Bt + (size_t)(bn * 64 + (tid & 63)) * K + k0 + 64, 0, 1);

#pragma unroll
    for (int kk = 0; kk < 64; kk += 32) {
      v16bf af[2], bfr[2], b2fr[2];
#pragma unroll
      for (int i = 0; i < 2; ++i) {
        const int r = wm * 32 + 16 * i + ln;
        af[i] = pack16(&As[buf][r][kk + kb16], &As[buf][r][kk + kb16 + 8]);
      }
#pragma unroll
      for (int j = 0; j < 2; ++j) {
        const size_t col = (size_t)(bn * 64 + wn * 32 + 16 * j + ln);
        bfr[j] = pack16(Bt + col * K + k0 + kk + kb16,
                        Bt + col * K + k0 + kk + kb16 + 8);
        if (EPI == 3)
          b2fr[j] = pack16(B2t + col * K + k0 + kk + kb16,
                           B2t + col * K + k0 + kk + kb16 + 8);
      }
#pragma unroll
      for (int i = 0; i < 2; ++i)
#pragma unroll
        for (int j = 0; j < 2; ++j) {
          acc[i][j] = wmma_bf16(af[i], bfr[j], acc[i][j]);
          if (EPI == 3) acc2[i][j] = wmma_bf16(af[i], b2fr[j], acc2[i][j]);
        }
    }
  }

  // --- epilogue (C layout: row = e + 8*(lane>>4), col = lane&15) ---
  const int r0 = bm * 128 + wm * 32;
  const int c0 = bn * 64 + wn * 32;
  const int rofs = 8 * (lane >> 4);
#pragma unroll
  for (int i = 0; i < 2; ++i)
#pragma unroll
    for (int j = 0; j < 2; ++j)
#pragma unroll
      for (int e = 0; e < 8; ++e) {
        const int r = r0 + 16 * i + e + rofs;
        const int c = c0 + 16 * j + ln;
        const size_t idx = (size_t)r * N + c;
        const float v = acc[i][j][e];
        if (EPI == 0) {
          outF[idx] = v;
        } else if (EPI == 2) {
          outF[idx] = v + res1[idx];
        } else if (EPI == 3) {
          const float gv = v, uv = acc2[i][j][e];
          const float s = gv / (1.f + __expf(-gv));
          outH[idx] = f2bf(s * uv);
        } else if (EPI == 4) {
          outF[idx] = v + res1[idx] + res2[idx];
        }
      }
}

// ---------------------------------------------------------------------------
// RoPE + pack to head-major bf16. Q is pre-scaled by 1/sqrt(HD) so the
// attention kernel needs no per-score multiply.
// ---------------------------------------------------------------------------
__global__ __launch_bounds__(1024) void rope_pack_kernel(
    const float* __restrict__ q, const float* __restrict__ k,
    const float* __restrict__ v, bf16_t* __restrict__ qb,
    bf16_t* __restrict__ kb, bf16_t* __restrict__ vb) {
  const int t = blockIdx.x;
  const int tid = threadIdx.x;
  const float LOG_THETA = 9.210340371976184f;  // ln(10000)

  {  // Q: 32 heads * 32 pairs, scaled by 0.125
    const int h = tid >> 5, i = tid & 31;
    const float inv = __expf(-(2.0f * i / 64.0f) * LOG_THETA);
    const float f = (float)t * inv;
    const float c = __cosf(f), s = __sinf(f);
    const float* qr = q + (size_t)t * D_MODEL + h * HD;
    const float x1 = qr[2 * i], x2 = qr[2 * i + 1];
    bf16_t* dst = qb + ((size_t)h * T_SEQ + t) * HD;
    dst[2 * i]     = f2bf((x1 * c - x2 * s) * 0.125f);
    dst[2 * i + 1] = f2bf((x2 * c + x1 * s) * 0.125f);
  }
  if (tid < 256) {  // K: 8 heads * 32 pairs
    const int h = tid >> 5, i = tid & 31;
    const float inv = __expf(-(2.0f * i / 64.0f) * LOG_THETA);
    const float f = (float)t * inv;
    const float c = __cosf(f), s = __sinf(f);
    const float* kr = k + (size_t)t * KV_DIM + h * HD;
    const float x1 = kr[2 * i], x2 = kr[2 * i + 1];
    bf16_t* dst = kb + ((size_t)h * T_SEQ + t) * HD;
    dst[2 * i]     = f2bf(x1 * c - x2 * s);
    dst[2 * i + 1] = f2bf(x2 * c + x1 * s);
  }
  if (tid < 512) {  // V: plain convert/pack
    const int h = tid >> 6, d = tid & 63;
    vb[((size_t)h * T_SEQ + t) * HD + d] = f2bf(v[(size_t)t * KV_DIM + h * HD + d]);
  }
}

// ---------------------------------------------------------------------------
// Flash attention (causal): one wave per 16-query tile, 32-key blocks.
// V tiles arrive via async global->LDS copies (ASYNCcnt); P transposes
// through per-wave LDS. No block-wide syncs (divergent causal trip counts).
// ---------------------------------------------------------------------------
__global__ __launch_bounds__(256) void flash_kernel(
    const bf16_t* __restrict__ qb, const bf16_t* __restrict__ kb,
    const bf16_t* __restrict__ vb, bf16_t* __restrict__ yb) {
  __shared__ __align__(128) bf16_t Plds[8][16][32];  // per-wave P tile
  __shared__ __align__(128) bf16_t Vrow[8][32][64];  // per-wave V tile (row-major)

  const int head = blockIdx.y, kvh = head >> 2;
  const int wave = threadIdx.x >> 5, lane = threadIdx.x & 31;
  const int ln = lane & 15, half = lane >> 4;
  const int kb8 = half * 8, kb16 = half * 16;
  const int qbase = blockIdx.x * 128 + wave * 16;

  // Q A-fragments (16 rows x 64 dims = two K=32 halves)
  const bf16_t* qrow = qb + ((size_t)head * T_SEQ + qbase + ln) * HD;
  const v16bf aq0 = pack16(qrow + kb16,      qrow + kb16 + 8);
  const v16bf aq1 = pack16(qrow + 32 + kb16, qrow + 32 + kb16 + 8);

  v8f o[4] = {};
  float m[8], l[8];
#pragma unroll
  for (int e = 0; e < 8; ++e) { m[e] = -3.0e30f; l[e] = 0.f; }

  const unsigned vlds = (unsigned)(size_t)&Vrow[wave][0][0];
  const int kend = qbase + 15;
  for (int kb0 = 0; kb0 <= kend; kb0 += 32) {
    // --- async-stage V block (32 tokens x 64 dims, 4KB) into this wave's LDS
    {
      const char* vbase =
          (const char*)(vb + ((size_t)kvh * T_SEQ + kb0) * HD);
#pragma unroll
      for (int i = 0; i < 8; ++i) {
        const unsigned off = (unsigned)(lane + 32 * i) * 16u;
        async_copy_b128(vlds + off, (unsigned long long)(vbase + off));
      }
      asm volatile("s_wait_asynccnt 0" ::: "memory");
    }

    // --- S = Q @ K^T for two 16-token subtiles ---
    v8f sc[2];
#pragma unroll
    for (int s = 0; s < 2; ++s) {
      const bf16_t* krow = kb + ((size_t)kvh * T_SEQ + kb0 + 16 * s + ln) * HD;
      const v16bf bk0 = pack16(krow + kb16,      krow + kb16 + 8);
      const v16bf bk1 = pack16(krow + 32 + kb16, krow + 32 + kb16 + 8);
      v8f c = {};
      c = wmma_bf16(aq0, bk0, c);
      c = wmma_bf16(aq1, bk1, c);
      sc[s] = c;
    }

    // --- causal mask (scores already scaled via Q) ---
#pragma unroll
    for (int s = 0; s < 2; ++s)
#pragma unroll
      for (int e = 0; e < 8; ++e) {
        const int qg = qbase + e + 8 * half;
        const int kg = kb0 + 16 * s + ln;
        if (kg > qg) sc[s][e] = -3.0e30f;
      }

    // --- online softmax: row max / exp / row sum (16-lane reductions) ---
    float alpha[8];
#pragma unroll
    for (int e = 0; e < 8; ++e) {
      float mx = fmaxf(sc[0][e], sc[1][e]);
#pragma unroll
      for (int msk = 1; msk < 16; msk <<= 1) mx = fmaxf(mx, __shfl_xor(mx, msk, 32));
      const float mn = fmaxf(m[e], mx);
      alpha[e] = __expf(m[e] - mn);
      m[e] = mn;
    }
#pragma unroll
    for (int e = 0; e < 8; ++e) {
      sc[0][e] = __expf(sc[0][e] - m[e]);
      sc[1][e] = __expf(sc[1][e] - m[e]);
      float rs = sc[0][e] + sc[1][e];
#pragma unroll
      for (int msk = 1; msk < 16; msk <<= 1) rs += __shfl_xor(rs, msk, 32);
      l[e] = l[e] * alpha[e] + rs;
#pragma unroll
      for (int t = 0; t < 4; ++t) o[t][e] *= alpha[e];
    }

    // --- P: C-layout -> A-layout via per-wave LDS round-trip ---
#pragma unroll
    for (int s = 0; s < 2; ++s)
#pragma unroll
      for (int e = 0; e < 8; ++e)
        Plds[wave][e + 8 * half][16 * s + ln] = f2bf(sc[s][e]);
    asm volatile("s_wait_dscnt 0" ::: "memory");
    const v16bf pa = pack16(&Plds[wave][ln][kb16], &Plds[wave][ln][kb16 + 8]);

    // --- O += P @ V (4 output dim-tiles); V B-fragments gathered from LDS ---
#pragma unroll
    for (int t = 0; t < 4; ++t) {
      v16bf bv;
#pragma unroll
      for (int j = 0; j < 16; ++j) bv[j] = Vrow[wave][kb16 + j][16 * t + ln];
      o[t] = wmma_bf16(pa, bv, o[t]);
    }
  }

  // --- normalize + write y[t][head*64 + d] (bf16) ---
#pragma unroll
  for (int t = 0; t < 4; ++t)
#pragma unroll
    for (int e = 0; e < 8; ++e) {
      const int qg = qbase + e + 8 * half;
      const int dim = 16 * t + ln;
      yb[(size_t)qg * D_MODEL + head * HD + dim] = f2bf(o[t][e] / l[e]);
    }
}

// ---------------------------------------------------------------------------
// Launch pipeline
// ---------------------------------------------------------------------------
extern "C" void kernel_launch(void* const* d_in, const int* in_sizes, int n_in,
                              void* d_out, int out_size, void* d_ws,
                              size_t ws_size, hipStream_t stream) {
  (void)in_sizes; (void)n_in; (void)out_size; (void)ws_size;
  const float* x  = (const float*)d_in[0];
  const float* g1 = (const float*)d_in[1];
  const float* wq = (const float*)d_in[2];
  const float* wk = (const float*)d_in[3];
  const float* wv = (const float*)d_in[4];
  const float* wo = (const float*)d_in[5];
  const float* g2 = (const float*)d_in[6];
  const float* wg = (const float*)d_in[7];
  const float* wu = (const float*)d_in[8];
  const float* wd = (const float*)d_in[9];
  float* out = (float*)d_out;

  char* p = (char*)d_ws;
  auto alloc = [&](size_t bytes) -> char* {
    char* r = p;
    p += (bytes + 255) & ~(size_t)255;
    return r;
  };
  bf16_t* xn_bf = (bf16_t*)alloc((size_t)T_SEQ * D_MODEL * 2);
  float*  xn_f  = (float*) alloc((size_t)T_SEQ * D_MODEL * 4);
  float*  qf    = (float*) alloc((size_t)T_SEQ * D_MODEL * 4);
  float*  kf    = (float*) alloc((size_t)T_SEQ * KV_DIM * 4);
  float*  vf    = (float*) alloc((size_t)T_SEQ * KV_DIM * 4);
  bf16_t* qbuf  = (bf16_t*)alloc((size_t)T_SEQ * D_MODEL * 2);
  bf16_t* kbuf  = (bf16_t*)alloc((size_t)T_SEQ * KV_DIM * 2);
  bf16_t* vbuf  = (bf16_t*)alloc((size_t)T_SEQ * KV_DIM * 2);
  bf16_t* ybuf  = (bf16_t*)alloc((size_t)T_SEQ * D_MODEL * 2);
  float*  x2    = (float*) alloc((size_t)T_SEQ * D_MODEL * 4);
  bf16_t* hbuf  = (bf16_t*)alloc((size_t)T_SEQ * FF_DIM * 2);
  // transposed bf16 weights: Wt[N][K]
  bf16_t* wqt = (bf16_t*)alloc((size_t)D_MODEL * D_MODEL * 2);
  bf16_t* wkt = (bf16_t*)alloc((size_t)KV_DIM * D_MODEL * 2);
  bf16_t* wvt = (bf16_t*)alloc((size_t)KV_DIM * D_MODEL * 2);
  bf16_t* wot = (bf16_t*)alloc((size_t)D_MODEL * D_MODEL * 2);
  bf16_t* wgt = (bf16_t*)alloc((size_t)FF_DIM * D_MODEL * 2);
  bf16_t* wut = (bf16_t*)alloc((size_t)FF_DIM * D_MODEL * 2);
  bf16_t* wdt = (bf16_t*)alloc((size_t)D_MODEL * FF_DIM * 2);

  // 0) one-pass weight convert+transpose to bf16 (bandwidth-trivial)
  convert_transpose_kernel<<<dim3(D_MODEL / 32, D_MODEL / 32), 256, 0, stream>>>(wq, wqt, D_MODEL, D_MODEL);
  convert_transpose_kernel<<<dim3(KV_DIM / 32, D_MODEL / 32), 256, 0, stream>>>(wk, wkt, D_MODEL, KV_DIM);
  convert_transpose_kernel<<<dim3(KV_DIM / 32, D_MODEL / 32), 256, 0, stream>>>(wv, wvt, D_MODEL, KV_DIM);
  convert_transpose_kernel<<<dim3(D_MODEL / 32, D_MODEL / 32), 256, 0, stream>>>(wo, wot, D_MODEL, D_MODEL);
  convert_transpose_kernel<<<dim3(FF_DIM / 32, D_MODEL / 32), 256, 0, stream>>>(wg, wgt, D_MODEL, FF_DIM);
  convert_transpose_kernel<<<dim3(FF_DIM / 32, D_MODEL / 32), 256, 0, stream>>>(wu, wut, D_MODEL, FF_DIM);
  convert_transpose_kernel<<<dim3(D_MODEL / 32, FF_DIM / 32), 256, 0, stream>>>(wd, wdt, FF_DIM, D_MODEL);

  // 1) xn1 = rms_norm(x, g1)
  rmsnorm_kernel<<<T_SEQ, 256, 0, stream>>>(x, g1, xn_f, xn_bf, D_MODEL);
  // 2) Q/K/V projections
  gemm_kernel<0><<<dim3(D_MODEL / 64, T_SEQ / 128), 256, 0, stream>>>(
      xn_bf, wqt, nullptr, qf, nullptr, nullptr, nullptr, T_SEQ, D_MODEL, D_MODEL);
  gemm_kernel<0><<<dim3(KV_DIM / 64, T_SEQ / 128), 256, 0, stream>>>(
      xn_bf, wkt, nullptr, kf, nullptr, nullptr, nullptr, T_SEQ, KV_DIM, D_MODEL);
  gemm_kernel<0><<<dim3(KV_DIM / 64, T_SEQ / 128), 256, 0, stream>>>(
      xn_bf, wvt, nullptr, vf, nullptr, nullptr, nullptr, T_SEQ, KV_DIM, D_MODEL);
  // 3) RoPE + head-major bf16 pack (Q pre-scaled)
  rope_pack_kernel<<<T_SEQ, 1024, 0, stream>>>(qf, kf, vf, qbuf, kbuf, vbuf);
  // 4) flash attention -> y (bf16, [T][D])
  flash_kernel<<<dim3(T_SEQ / 128, N_HEAD), 256, 0, stream>>>(qbuf, kbuf, vbuf, ybuf);
  // 5) x2 = xn1 + y @ wo
  gemm_kernel<2><<<dim3(D_MODEL / 64, T_SEQ / 128), 256, 0, stream>>>(
      ybuf, wot, nullptr, x2, nullptr, xn_f, nullptr, T_SEQ, D_MODEL, D_MODEL);
  // 6) xn2 = rms_norm(x2, g2) (reuse xn buffers)
  rmsnorm_kernel<<<T_SEQ, 256, 0, stream>>>(x2, g2, xn_f, xn_bf, D_MODEL);
  // 7) h = silu(xn2 @ wg) * (xn2 @ wu)  (fused dual-B GEMM)
  gemm_kernel<3><<<dim3(FF_DIM / 64, T_SEQ / 128), 256, 0, stream>>>(
      xn_bf, wgt, wut, nullptr, hbuf, nullptr, nullptr, T_SEQ, FF_DIM, D_MODEL);
  // 8) out = xn2 + h @ wd + x
  gemm_kernel<4><<<dim3(D_MODEL / 64, T_SEQ / 128), 256, 0, stream>>>(
      hbuf, wdt, nullptr, out, nullptr, xn_f, x, T_SEQ, D_MODEL, FF_DIM);
}
